// TransformerEncoder_26895085208033
// MI455X (gfx1250) — matrix operand order repaired and checked by reading, not compiled, not run on hardware
//
#include <hip/hip_runtime.h>

// ---- model dims (fixed by reference) ----
#define L_  6
#define B_  2
#define S_  2048
#define D_  1024
#define H_  16
#define DK_ 64
#define F_  4096
#define M_  (B_*S_)   // 4096 token rows

typedef __attribute__((ext_vector_type(16))) __bf16 v16bf;
typedef __attribute__((ext_vector_type(8)))  __bf16 v8bf;
typedef __attribute__((ext_vector_type(8)))  float  v8f;

static __device__ inline v8f wmma_bf16(v16bf a, v16bf b, v8f c) {
  return __builtin_amdgcn_wmma_f32_16x16x32_bf16(false, a, false, b, (short)0, c, false, false);
}
static __device__ inline v8f zero_v8f() {
  v8f z;
#pragma unroll
  for (int g = 0; g < 8; ++g) z[g] = 0.0f;
  return z;
}

// ---- CDNA5 async global->LDS DMA (ASYNCcnt-tracked, no VGPR round trip) ----
static __device__ inline unsigned lds_off(const void* p) {
  // flat LDS aperture keeps the wave-relative LDS byte offset in addr[31:0]
  return (unsigned)(unsigned long long)p;
}
static __device__ inline void async_ld_b128(unsigned lds, const void* g) {
  asm volatile("global_load_async_to_lds_b128 %0, %1, off"
               :: "v"(lds), "v"(g) : "memory");
}
static __device__ inline void async_wait0() {
  asm volatile("s_wait_asynccnt 0x0" ::: "memory");
}
// copy 64 contiguous bytes (32 bf16) global -> LDS
static __device__ inline void async_copy64B(void* lds_p, const __bf16* g) {
  unsigned l = lds_off(lds_p);
  async_ld_b128(l,      g);
  async_ld_b128(l + 16, g + 8);
  async_ld_b128(l + 32, g + 16);
  async_ld_b128(l + 48, g + 24);
}

// ---------------- transposing f32 -> bf16 weight conversion ----------------
// in: [L][K][N] f32  ->  out: [L][N][K] bf16 (so GEMM B-tiles are K-contiguous)
__global__ __launch_bounds__(256) void cvt_t_kernel(const float* __restrict__ in,
                                                    __bf16* __restrict__ out,
                                                    int K, int N) {
  __shared__ float t[32][33];                        // padded: no bank conflicts
  const size_t plane = (size_t)K * N;
  const float* ip = in + (size_t)blockIdx.z * plane;
  __bf16* op = out + (size_t)blockIdx.z * plane;
  int n0 = blockIdx.x * 32, k0 = blockIdx.y * 32;
  int tx = threadIdx.x & 31, ty = threadIdx.x >> 5;  // 32 x 8
#pragma unroll
  for (int i = ty; i < 32; i += 8) t[i][tx] = ip[(size_t)(k0 + i) * N + n0 + tx];
  __syncthreads();
#pragma unroll
  for (int i = ty; i < 32; i += 8) op[(size_t)(n0 + i) * K + k0 + tx] = (__bf16)t[tx][i];
}

// ---------------- embedding + positional encoding ----------------
__global__ __launch_bounds__(256) void embed_kernel(const int* __restrict__ tokens,
                                                    const float* __restrict__ emb,
                                                    float* __restrict__ x) {
  int idx = blockIdx.x * 256 + threadIdx.x;          // over M_*D_
  int m = idx / D_, d = idx % D_;
  int tok = tokens[m];
  int s = m % S_;
  const float scale = 45.254833995939045f;           // sqrt(2048): ref scales by sqrt(seq_len)
  int i2 = d >> 1;
  float div = __expf(-(float)(2 * i2) * (9.210340371976184f / (float)D_)); // ln(10000)
  float ang = (float)s * div;
  float pe = (d & 1) ? __cosf(ang) : __sinf(ang);
  x[idx] = emb[(size_t)tok * D_ + d] * scale + pe;
}

// ---------------- LayerNorm ----------------
template <int OUTBF>
__global__ __launch_bounds__(256) void ln_kernel(const float* __restrict__ x,
                                                 const float* __restrict__ gamma,
                                                 const float* __restrict__ beta,
                                                 __bf16* __restrict__ outb,
                                                 float* __restrict__ outf) {
  __shared__ float sh_s[8], sh_s2[8];
  int row = blockIdx.x;
  const float* xr = x + (size_t)row * D_;
  float s = 0.f, s2 = 0.f;
  for (int d = threadIdx.x; d < D_; d += 256) { float v = xr[d]; s += v; s2 += v * v; }
#pragma unroll
  for (int off = 16; off > 0; off >>= 1) { s += __shfl_xor(s, off, 32); s2 += __shfl_xor(s2, off, 32); }
  int wid = threadIdx.x >> 5, lane = threadIdx.x & 31;
  if (lane == 0) { sh_s[wid] = s; sh_s2[wid] = s2; }
  __syncthreads();
  float ts = 0.f, t2 = 0.f;
#pragma unroll
  for (int i = 0; i < 8; ++i) { ts += sh_s[i]; t2 += sh_s2[i]; }
  float mean = ts * (1.0f / D_);
  float var = t2 * (1.0f / D_) - mean * mean;
  float rstd = rsqrtf(var + 1e-5f);
  for (int d = threadIdx.x; d < D_; d += 256) {
    float v = (xr[d] - mean) * rstd * gamma[d] + beta[d];
    if (OUTBF) outb[(size_t)row * D_ + d] = (__bf16)v;
    else       outf[(size_t)row * D_ + d] = v;
  }
}

// ---------------- tiled bf16 WMMA GEMM, fully-async double-buffered ----------------
// C(M_,N) = A(M_,K) @ Wt(N,K)^T + bias     (Wt is pre-transposed, [N][K])
// MODE 0: out_bf[row][col] = C
// MODE 1: out_bf[row][col] = relu(C)
// MODE 2: out_f[row][col] += C             (f32 residual add)
// MODE 3: out_bf[col * M_ + row] = C        (transposed store, for V^T)
template <int MODE>
__global__ __launch_bounds__(256) void gemm_kernel(const __bf16* __restrict__ A,
                                                   const __bf16* __restrict__ W,
                                                   const float* __restrict__ bias,
                                                   __bf16* __restrict__ outb,
                                                   float* __restrict__ outf,
                                                   int N, int K) {
  __shared__ __align__(32) __bf16 sA[2][128 * 64];    // [m][k]
  __shared__ __align__(32) __bf16 sBt[2][128 * 64];   // [n][k]
  const int tid = threadIdx.x;
  const int lane = tid & 31, wid = tid >> 5;
  const int wm = wid & 1, wn = wid >> 1;              // 2x4 wave grid over 128x128
  const int lr = lane & 15, half = lane >> 4;
  const int row0 = blockIdx.y * 128;
  const int col0 = blockIdx.x * 128;

  v8f acc[4][2];
#pragma unroll
  for (int mt = 0; mt < 4; ++mt)
#pragma unroll
    for (int nt = 0; nt < 2; ++nt) acc[mt][nt] = zero_v8f();

  auto stage = [&](int kb, int buf) {                 // 100% async DMA staging
    int r = tid >> 1, seg = tid & 1;
    async_copy64B(&sA[buf][r * 64 + seg * 32],
                  A + (size_t)(row0 + r) * K + kb + seg * 32);
    async_copy64B(&sBt[buf][r * 64 + seg * 32],
                  W + (size_t)(col0 + r) * K + kb + seg * 32);
  };

  const int nk = K / 64;
  stage(0, 0);
  for (int it = 0; it < nk; ++it) {
    const int buf = it & 1;
    async_wait0();          // our async LDS writes complete
    __syncthreads();        // tile `it` visible workgroup-wide
    if (it + 1 < nk) stage((it + 1) * 64, buf ^ 1);  // overlap DMA with compute

    const __bf16* cA = sA[buf];
    const __bf16* cB = sBt[buf];
#pragma unroll
    for (int kk = 0; kk < 64; kk += 32) {
      v16bf af[4], bf2[2];
#pragma unroll
      for (int mt = 0; mt < 4; ++mt) {   // A-layout: lanes<16 K0..15; lanes>=16 K16..31
        const __bf16* ap = cA + (wm * 64 + mt * 16 + lr) * 64 + kk + half * 16;
        v16bf a;
#pragma unroll
        for (int e = 0; e < 8; ++e) { a[e] = ap[e]; a[e + 8] = ap[e + 8]; }
        af[mt] = a;
      }
#pragma unroll
      for (int nt = 0; nt < 2; ++nt) {   // B-layout: lanes<16 K0..15; lanes>=16 K16..31 (32B contig)
        const __bf16* bp = cB + (wn * 32 + nt * 16 + lr) * 64 + kk + half * 16;
        bf2[nt] = *(const v16bf*)bp;
      }
#pragma unroll
      for (int mt = 0; mt < 4; ++mt)
#pragma unroll
        for (int nt = 0; nt < 2; ++nt)
          acc[mt][nt] = wmma_bf16(af[mt], bf2[nt], acc[mt][nt]);
    }
    __syncthreads();        // everyone done reading buf before it is re-staged
  }

  // epilogue (C layout: VGPR g -> row g (lanes 0-15) / g+8 (lanes 16-31), col = lane%16)
#pragma unroll
  for (int mt = 0; mt < 4; ++mt)
#pragma unroll
    for (int nt = 0; nt < 2; ++nt) {
      int col = col0 + wn * 32 + nt * 16 + lr;
      float bv = bias[col];
      if constexpr (MODE == 3) {
        // 8 accumulator elements are 8 consecutive rows -> one contiguous 16B store
        int rowb = row0 + wm * 64 + mt * 16 + half * 8;
        v8bf pk;
#pragma unroll
        for (int g = 0; g < 8; ++g) pk[g] = (__bf16)(acc[mt][nt][g] + bv);
        *(v8bf*)(outb + (size_t)col * M_ + rowb) = pk;
      } else {
#pragma unroll
        for (int g = 0; g < 8; ++g) {
          int row = row0 + wm * 64 + mt * 16 + g + half * 8;
          float v = acc[mt][nt][g] + bv;
          if constexpr (MODE == 0)      outb[(size_t)row * N + col] = (__bf16)v;
          else if constexpr (MODE == 1) outb[(size_t)row * N + col] = (__bf16)fmaxf(v, 0.0f);
          else                          outf[(size_t)row * N + col] += v;
        }
      }
    }
}

// ---------------- flash attention (online softmax, bf16 WMMA) ----------------
// grid: (S/64, H, B), 128 threads (4 waves, 16 query rows each)
// Vt is the transposed V: [D][M_] bf16 (produced by GEMM MODE 3)
__global__ __launch_bounds__(128) void attn_kernel(const __bf16* __restrict__ Q,
                                                   const __bf16* __restrict__ Kb,
                                                   const __bf16* __restrict__ Vt,
                                                   const int* __restrict__ maskp,
                                                   __bf16* __restrict__ O) {
  __shared__ __align__(32) __bf16 sK[64 * 64];       // [key][d]
  __shared__ __align__(32) __bf16 sVt[64 * 64];      // [d][key]
  __shared__ __align__(32) __bf16 sP[4 * 16 * 64];   // per-wave P tile [16][64]
  const int tid = threadIdx.x, lane = tid & 31, w = tid >> 5;
  const int lr = lane & 15, half = lane >> 4;
  const int qb = blockIdx.x, h = blockIdx.y, bz = blockIdx.z;
  const int hoff = h * DK_;
  const size_t base = (size_t)bz * S_ * D_;

  // preload Q fragments (16 rows x 64 k, two K=32 steps)
  v16bf aq[2];
  {
    int qrow = qb * 64 + w * 16 + lr;
    const __bf16* qp = Q + base + (size_t)qrow * D_ + hoff;
#pragma unroll
    for (int kk = 0; kk < 2; ++kk) {
      const __bf16* p = qp + kk * 32 + half * 16;
      v16bf a;
#pragma unroll
      for (int e = 0; e < 8; ++e) { a[e] = p[e]; a[e + 8] = p[e + 8]; }
      aq[kk] = a;
    }
  }

  float mrow[8], lrow[8];
  v8f accO[4];
#pragma unroll
  for (int g = 0; g < 8; ++g) { mrow[g] = -1e30f; lrow[g] = 0.0f; }
#pragma unroll
  for (int dt = 0; dt < 4; ++dt) accO[dt] = zero_v8f();

  const float scl = 0.125f;  // 1/sqrt(DK)

  for (int kb2 = 0; kb2 < S_ / 64; ++kb2) {
    { // both tiles staged via async DMA
      int r = tid >> 1, seg = tid & 1;
      async_copy64B(&sK[r * 64 + seg * 32],
                    Kb + base + (size_t)(kb2 * 64 + r) * D_ + hoff + seg * 32);
      async_copy64B(&sVt[r * 64 + seg * 32],
                    Vt + (size_t)(hoff + r) * M_ + bz * S_ + kb2 * 64 + seg * 32);
    }
    async_wait0();
    __syncthreads();

    // S = Q @ K^T : 4 key-subtiles of 16, two K=32 steps each
    v8f sacc[4];
#pragma unroll
    for (int nt = 0; nt < 4; ++nt) sacc[nt] = zero_v8f();
#pragma unroll
    for (int nt = 0; nt < 4; ++nt)
#pragma unroll
      for (int kk = 0; kk < 2; ++kk) {
        const __bf16* bp = sK + (nt * 16 + lr) * 64 + kk * 32 + half * 16;  // contiguous 32B
        v16bf bk = *(const v16bf*)bp;
        sacc[nt] = wmma_bf16(aq[kk], bk, sacc[nt]);
      }

    int mk[4];
#pragma unroll
    for (int nt = 0; nt < 4; ++nt) mk[nt] = maskp[bz * S_ + kb2 * 64 + nt * 16 + lr];

    // online softmax per row (row = g + half*8 in C layout; 16-lane groups share a row)
#pragma unroll
    for (int g = 0; g < 8; ++g) {
      float sv[4];
#pragma unroll
      for (int nt = 0; nt < 4; ++nt) {
        float sc = sacc[nt][g] * scl;
        sv[nt] = mk[nt] ? sc : -1e9f;
      }
      float tmax = fmaxf(fmaxf(sv[0], sv[1]), fmaxf(sv[2], sv[3]));
#pragma unroll
      for (int off = 8; off > 0; off >>= 1) tmax = fmaxf(tmax, __shfl_xor(tmax, off, 32));
      float nm = fmaxf(mrow[g], tmax);
      float alpha = __expf(mrow[g] - nm);
      float ps = 0.0f;
      int r = g + half * 8;
#pragma unroll
      for (int nt = 0; nt < 4; ++nt) {
        float p = __expf(sv[nt] - nm);
        ps += p;
        sP[w * 1024 + r * 64 + nt * 16 + lr] = (__bf16)p;  // C layout -> row-major P in LDS
      }
#pragma unroll
      for (int off = 8; off > 0; off >>= 1) ps += __shfl_xor(ps, off, 32);
      lrow[g] = lrow[g] * alpha + ps;
      mrow[g] = nm;
#pragma unroll
      for (int dt = 0; dt < 4; ++dt) accO[dt][g] *= alpha;
    }

    // O += P @ V  (P re-read from per-wave LDS in A layout; V from transposed tile)
#pragma unroll
    for (int kk = 0; kk < 2; ++kk) {
      const __bf16* pp = sP + w * 1024 + lr * 64 + kk * 32 + half * 16;
      v16bf ap;
#pragma unroll
      for (int e = 0; e < 8; ++e) { ap[e] = pp[e]; ap[e + 8] = pp[e + 8]; }
#pragma unroll
      for (int dt = 0; dt < 4; ++dt) {
        const __bf16* vp = sVt + (dt * 16 + lr) * 64 + kk * 32 + half * 16;  // contiguous 32B
        v16bf bv = *(const v16bf*)vp;
        accO[dt] = wmma_bf16(ap, bv, accO[dt]);
      }
    }
    __syncthreads();
  }

  // write attn output (bf16, consumed by O-projection GEMM)
#pragma unroll
  for (int dt = 0; dt < 4; ++dt)
#pragma unroll
    for (int g = 0; g < 8; ++g) {
      int row = qb * 64 + w * 16 + g + half * 8;
      float v = accO[dt][g] / lrow[g];
      O[base + (size_t)row * D_ + hoff + dt * 16 + lr] = (__bf16)v;
    }
}

// ---------------- host-side orchestration ----------------
extern "C" void kernel_launch(void* const* d_in, const int* in_sizes, int n_in,
                              void* d_out, int out_size, void* d_ws, size_t ws_size,
                              hipStream_t stream) {
  (void)in_sizes; (void)n_in; (void)out_size; (void)ws_size;
  const int*   tokens = (const int*)d_in[0];
  const int*   maskp  = (const int*)d_in[1];
  const float* emb = (const float*)d_in[2];
  const float* Wq  = (const float*)d_in[3];
  const float* bq  = (const float*)d_in[4];
  const float* Wk  = (const float*)d_in[5];
  const float* bk  = (const float*)d_in[6];
  const float* Wv  = (const float*)d_in[7];
  const float* bv  = (const float*)d_in[8];
  const float* Wo  = (const float*)d_in[9];
  const float* bo  = (const float*)d_in[10];
  const float* W1  = (const float*)d_in[11];
  const float* b1  = (const float*)d_in[12];
  const float* W2  = (const float*)d_in[13];
  const float* b2  = (const float*)d_in[14];
  const float* g1  = (const float*)d_in[15];
  const float* be1 = (const float*)d_in[16];
  const float* g2  = (const float*)d_in[17];
  const float* be2 = (const float*)d_in[18];
  const float* gf  = (const float*)d_in[19];
  const float* bf  = (const float*)d_in[20];
  float* out = (float*)d_out;

  char* ws = (char*)d_ws;
  size_t off = 0;
  auto alloc = [&](size_t bytes) -> void* {
    void* p = ws + off;
    off += (bytes + 255) & ~(size_t)255;
    return p;
  };
  const size_t LDD = (size_t)L_ * D_ * D_;
  const size_t LDF = (size_t)L_ * D_ * F_;
  __bf16* wqb = (__bf16*)alloc(LDD * 2);   // all weights stored transposed: [L][N][K]
  __bf16* wkb = (__bf16*)alloc(LDD * 2);
  __bf16* wvb = (__bf16*)alloc(LDD * 2);
  __bf16* wob = (__bf16*)alloc(LDD * 2);
  __bf16* w1b = (__bf16*)alloc(LDF * 2);
  __bf16* w2b = (__bf16*)alloc(LDF * 2);
  float*  x    = (float*) alloc((size_t)M_ * D_ * 4);
  __bf16* hbf  = (__bf16*)alloc((size_t)M_ * D_ * 2);
  __bf16* qbf  = (__bf16*)alloc((size_t)M_ * D_ * 2);
  __bf16* kbf  = (__bf16*)alloc((size_t)M_ * D_ * 2);
  __bf16* vtb  = (__bf16*)alloc((size_t)D_ * M_ * 2);   // V^T: [D][M_]
  __bf16* abf  = (__bf16*)alloc((size_t)M_ * D_ * 2);
  __bf16* fbf  = (__bf16*)alloc((size_t)M_ * F_ * 2);

  { // transposing weight conversion: [K][N] f32 -> [N][K] bf16, per layer in grid.z
    dim3 gDD(D_ / 32, D_ / 32, L_);
    cvt_t_kernel<<<gDD, 256, 0, stream>>>(Wq, wqb, D_, D_);
    cvt_t_kernel<<<gDD, 256, 0, stream>>>(Wk, wkb, D_, D_);
    cvt_t_kernel<<<gDD, 256, 0, stream>>>(Wv, wvb, D_, D_);
    cvt_t_kernel<<<gDD, 256, 0, stream>>>(Wo, wob, D_, D_);
    dim3 gDF(F_ / 32, D_ / 32, L_);   // W1: K=D, N=F
    cvt_t_kernel<<<gDF, 256, 0, stream>>>(W1, w1b, D_, F_);
    dim3 gFD(D_ / 32, F_ / 32, L_);   // W2: K=F, N=D
    cvt_t_kernel<<<gFD, 256, 0, stream>>>(W2, w2b, F_, D_);
  }
  embed_kernel<<<(M_ * D_) / 256, 256, 0, stream>>>(tokens, emb, x);

  const dim3 gQKV(D_ / 128, M_ / 128);   // (8,32)
  const dim3 gF1 (F_ / 128, M_ / 128);   // (32,32)
  const dim3 gF2 (D_ / 128, M_ / 128);
  const dim3 gAtt(S_ / 64, H_, B_);

  for (int i = 0; i < L_; ++i) {
    const __bf16* wq_i = wqb + (size_t)i * D_ * D_;
    const __bf16* wk_i = wkb + (size_t)i * D_ * D_;
    const __bf16* wv_i = wvb + (size_t)i * D_ * D_;
    const __bf16* wo_i = wob + (size_t)i * D_ * D_;
    const __bf16* w1_i = w1b + (size_t)i * D_ * F_;
    const __bf16* w2_i = w2b + (size_t)i * F_ * D_;

    ln_kernel<1><<<M_, 256, 0, stream>>>(x, g1 + (size_t)i * D_, be1 + (size_t)i * D_, hbf, nullptr);
    gemm_kernel<0><<<gQKV, 256, 0, stream>>>(hbf, wq_i, bq + (size_t)i * D_, qbf, nullptr, D_, D_);
    gemm_kernel<0><<<gQKV, 256, 0, stream>>>(hbf, wk_i, bk + (size_t)i * D_, kbf, nullptr, D_, D_);
    gemm_kernel<3><<<gQKV, 256, 0, stream>>>(hbf, wv_i, bv + (size_t)i * D_, vtb, nullptr, D_, D_);
    attn_kernel<<<gAtt, 128, 0, stream>>>(qbf, kbf, vtb, maskp, abf);
    gemm_kernel<2><<<gQKV, 256, 0, stream>>>(abf, wo_i, bo + (size_t)i * D_, nullptr, x, D_, D_);
    ln_kernel<1><<<M_, 256, 0, stream>>>(x, g2 + (size_t)i * D_, be2 + (size_t)i * D_, hbf, nullptr);
    gemm_kernel<1><<<gF1, 256, 0, stream>>>(hbf, w1_i, b1 + (size_t)i * F_, fbf, nullptr, F_, D_);
    gemm_kernel<2><<<gF2, 256, 0, stream>>>(fbf, w2_i, b2 + (size_t)i * D_, nullptr, x, D_, F_);
  }
  ln_kernel<0><<<M_, 256, 0, stream>>>(x, gf, bf, nullptr, out);
}
